// MobiusGRUCell_52432960749618
// MI455X (gfx1250) — matrix operand order repaired and checked by reading, not compiled, not run on hardware
//
#include <hip/hip_runtime.h>

typedef __attribute__((ext_vector_type(16))) __bf16    v16bf;
typedef __attribute__((ext_vector_type(8)))  float     v8f;
typedef __attribute__((ext_vector_type(4)))  unsigned  v4u;

union AB { v4u u[2]; v16bf v; };

#define APITCH 264   // LDS row pitch (bf16 elems): 528B -> bank stride 4, avoids 16-way conflicts

__device__ __forceinline__ float wave_sum(float v){
#pragma unroll
  for (int off = 16; off > 0; off >>= 1) v += __shfl_xor(v, off, 32);
  return v;
}
__device__ __forceinline__ float artanh_pos(float x){           // x >= 0
  x = fminf(x, 1.0f - 1e-7f);
  return 0.5f * logf((1.0f + x) / (1.0f - x));
}
__device__ __forceinline__ float sigmoidf(float x){ return 1.0f / (1.0f + expf(-x)); }
__device__ __forceinline__ unsigned short f2bf(float f){
  unsigned u = __float_as_uint(f);
  u += 0x7FFFu + ((u >> 16) & 1u);                              // round-to-nearest-even
  return (unsigned short)(u >> 16);
}

// Convert 16 rows x 256 cols of f32 -> bf16 into LDS (pitch APITCH), 256 threads.
__device__ __forceinline__ void stage_bf16(const float* __restrict__ src,
                                           unsigned short* dst, int tid)
{
  const int m   = tid >> 4;            // local row 0..15
  const int col = (tid & 15) * 16;     // 16 elems per thread
  const float4* g = (const float4*)(src + m*256 + col);
  union { unsigned short s[8]; v4u u; } a, b;
#pragma unroll
  for (int q = 0; q < 2; ++q){
    float4 f = g[q];
    a.s[q*4+0] = f2bf(f.x); a.s[q*4+1] = f2bf(f.y);
    a.s[q*4+2] = f2bf(f.z); a.s[q*4+3] = f2bf(f.w);
  }
#pragma unroll
  for (int q = 0; q < 2; ++q){
    float4 f = g[2+q];
    b.s[q*4+0] = f2bf(f.x); b.s[q*4+1] = f2bf(f.y);
    b.s[q*4+2] = f2bf(f.z); b.s[q*4+3] = f2bf(f.w);
  }
  *(v4u*)(dst + m*APITCH + col)     = a.u;
  *(v4u*)(dst + m*APITCH + col + 8) = b.u;
}

// Per-row fused: d = mobius_add(mobius_add(matvec, matvec), bias); y = LN(logmap0(d)).
// mh/mx: this row's 256-wide GEMM outputs in LDS; nh/nx: norms of matvec'd vectors;
// b2/bsum: sum(bias^2), sum(bias). Lane handles cols lane+32j.
__device__ __forceinline__ void gate_ln(
    const float* mh, const float* mx,
    const float* __restrict__ bias, const float* __restrict__ gam,
    const float* __restrict__ bet,
    float nh, float nx, float b2, float bsum, int lane, float y[8])
{
  float vh[8], vx[8], vb[8];
  float s0=0,s1=0,s2=0,s3=0,s4=0,s5=0,s6=0;
#pragma unroll
  for (int j = 0; j < 8; ++j){
    int c = lane + 32*j;
    vh[j] = mh[c]; vx[j] = mx[c]; vb[j] = bias[c];
    s0 += vh[j]*vh[j]; s1 += vx[j]*vx[j]; s2 += vh[j]*vx[j];
    s3 += vh[j]*vb[j]; s4 += vx[j]*vb[j]; s5 += vh[j]; s6 += vx[j];
  }
  float mh2 = wave_sum(s0), mx2 = wave_sum(s1), mhmx = wave_sum(s2);
  float mhb = wave_sum(s3), mxb = wave_sum(s4);
  float smh = wave_sum(s5), smx = wave_sum(s6);

  float nh_c = fmaxf(nh, 1e-7f), nx_c = fmaxf(nx, 1e-7f);
  float mhn_r = sqrtf(mh2), mxn_r = sqrtf(mx2);
  float mhn = fmaxf(mhn_r, 1e-7f), mxn = fmaxf(mxn_r, 1e-7f);
  // mobius_matvec scaling factors (zero-guard per reference where())
  float ah = (mhn_r <= 1e-7f) ? 0.f : tanhf(mhn/nh_c * artanh_pos(nh_c)) / mhn;
  float ax = (mxn_r <= 1e-7f) ? 0.f : tanhf(mxn/nx_c * artanh_pos(nx_c)) / mxn;
  // mobius_add(Wh, Ux)
  float a2 = ah*ah*mh2, bb2 = ax*ax*mx2, ab = ah*ax*mhmx;
  float den1 = fmaxf(1.f + 2.f*ab + a2*bb2, 1e-15f);
  float u1 = (1.f + 2.f*ab + bb2)/den1, v1 = (1.f - a2)/den1;
  float P0 = u1*ah, Q0 = v1*ax;
  float c2 = u1*u1*a2 + 2.f*u1*v1*ab + v1*v1*bb2;
  float cb = P0*mhb + Q0*mxb;
  // mobius_add(c, bias)
  float den2 = fmaxf(1.f + 2.f*cb + c2*b2, 1e-15f);
  float u2 = (1.f + 2.f*cb + b2)/den2, v2 = (1.f - c2)/den2;
  float P = u2*P0, Q = u2*Q0, S = v2;
  // logmap0 + layernorm in closed form
  float dn2 = u2*u2*c2 + 2.f*u2*v2*cb + v2*v2*b2;
  float dn = fmaxf(sqrtf(dn2), 1e-7f);
  float at = artanh_pos(dn);
  float cl = at/dn;
  float mu = cl * (P*smh + Q*smx + S*bsum) * (1.f/256.f);
  float var = at*at*(1.f/256.f) - mu*mu;
  float rstd = rsqrtf(var + 1e-5f);
#pragma unroll
  for (int j = 0; j < 8; ++j){
    int c = lane + 32*j;
    float d = P*vh[j] + Q*vx[j] + S*vb[j];
    y[j] = (cl*d - mu)*rstd*gam[c] + bet[c];
  }
}

__global__ void k_cvt(const float* __restrict__ in, unsigned short* __restrict__ out, long n){
  long i = (long)blockIdx.x*blockDim.x + threadIdx.x;
  long stride = (long)gridDim.x*blockDim.x;
  for (; i < n; i += stride) out[i] = f2bf(in[i]);
}

#define WMMA_BF16(A, Bm, C) \
  __builtin_amdgcn_wmma_f32_16x16x32_bf16(false, (A).v, false, (Bm).v, (short)0, (C), false, false)

// ---------------- Single fused Mobius-GRU kernel ---------------------------
// One 256-thread workgroup owns 16 rows end-to-end. Phase A: 5 WMMA GEMMs
// (mh_r, mh_z, mx_r, mx_z from LDS-staged bf16 hx/x; mx_h kept in VGPRs).
// Epilogue A -> z_t, rh_t in registers; rh_t re-staged in LDS as bf16 A-matrix.
// Phase B: mh_h = rh @ W_hh_^T. Epilogue B -> out. HBM: read x,hx once, write out.
__global__ __launch_bounds__(256) void k_mobius_gru(
    const unsigned short* __restrict__ wih, const unsigned short* __restrict__ whh,
    const float* __restrict__ xf, const float* __restrict__ hf,
    const float* __restrict__ bias, const float* __restrict__ gam, const float* __restrict__ bet,
    float* __restrict__ out)
{
  __shared__ float smem[4*16*256];                 // 64 KB, multi-phase reuse
  unsigned short* sAh = (unsigned short*)smem;     // phase A staging (dead before spill)
  unsigned short* sAx = sAh + 16*APITCH;
  const int tid  = threadIdx.x;
  const int wave = tid >> 5;
  const int lane = tid & 31;
  const int lr   = lane & 15;
  const int half = lane >> 4;
  const size_t rowbase = (size_t)blockIdx.x * 16;

  stage_bf16(hf + rowbase*256, sAh, tid);
  stage_bf16(xf + rowbase*256, sAx, tid);
  __syncthreads();

  // ---- Phase A: 5 GEMMs over K=256 ----
  v8f acc[4][2] = {};                              // mh_r, mh_z, mx_r, mx_z
  v8f accH[2]  = {};                               // mx_h = x @ W_ih_^T (stays in VGPRs)
  const unsigned short* rowAh = sAh + lr*APITCH;
  const unsigned short* rowAx = sAx + lr*APITCH;

  for (int kb = 0; kb < 256; kb += 32){
    AB Ah, Ax;
    const int ka = kb + half*16;                   // A frag: 16 contiguous K per lane
    Ah.u[0] = *(const v4u*)(rowAh + ka); Ah.u[1] = *(const v4u*)(rowAh + ka + 8);
    Ax.u[0] = *(const v4u*)(rowAx + ka); Ax.u[1] = *(const v4u*)(rowAx + ka + 8);
    const int kB = kb + half*16;                   // B frag: 16 contiguous K per lane
#pragma unroll
    for (int t = 0; t < 2; ++t){
      const int n = wave*32 + t*16 + lr;
      AB Bhr, Bhz, Bir, Biz, Bih; const unsigned short* p;
      p = whh + (size_t)(      n)*256 + kB; Bhr.u[0]=*(const v4u*)p; Bhr.u[1]=*(const v4u*)(p+8);
      p = whh + (size_t)(512 + n)*256 + kB; Bhz.u[0]=*(const v4u*)p; Bhz.u[1]=*(const v4u*)(p+8);
      p = wih + (size_t)(      n)*256 + kB; Bir.u[0]=*(const v4u*)p; Bir.u[1]=*(const v4u*)(p+8);
      p = wih + (size_t)(512 + n)*256 + kB; Biz.u[0]=*(const v4u*)p; Biz.u[1]=*(const v4u*)(p+8);
      p = wih + (size_t)(256 + n)*256 + kB; Bih.u[0]=*(const v4u*)p; Bih.u[1]=*(const v4u*)(p+8);
      acc[0][t] = WMMA_BF16(Ah, Bhr, acc[0][t]);   // hx @ W_hr^T
      acc[1][t] = WMMA_BF16(Ah, Bhz, acc[1][t]);   // hx @ W_hz^T
      acc[2][t] = WMMA_BF16(Ax, Bir, acc[2][t]);   //  x @ W_ir^T
      acc[3][t] = WMMA_BF16(Ax, Biz, acc[3][t]);   //  x @ W_iz^T
      accH[t]   = WMMA_BF16(Ax, Bih, accH[t]);     //  x @ W_ih_^T
    }
  }
  __syncthreads();                                 // staging dead; reuse LDS for spills
#pragma unroll
  for (int g = 0; g < 4; ++g)
#pragma unroll
    for (int t = 0; t < 2; ++t)
#pragma unroll
      for (int v = 0; v < 8; ++v)
        smem[g*4096 + (half*8 + v)*256 + wave*32 + t*16 + lr] = acc[g][t][v];
  __syncthreads();

  // ---- Epilogue A: z_t, r_t, rh_t (registers) ----
  float bz2, bzs, br2, brs;
  { float p2=0, ps=0, q2=0, qs=0;
#pragma unroll
    for (int j = 0; j < 8; ++j){ int c = lane + 32*j;
      float bz = bias[512 + c]; p2 += bz*bz; ps += bz;
      float br = bias[c];       q2 += br*br; qs += br; }
    bz2 = wave_sum(p2); bzs = wave_sum(ps);
    br2 = wave_sum(q2); brs = wave_sum(qs);
  }

  float zv[2][8], hxv[2][8], rhv[2][8];
  float hnv[2], xnv[2], rhn_[2];
  for (int ri = 0; ri < 2; ++ri){                  // each wave: rows 2w, 2w+1
    const int m = wave*2 + ri;
    const size_t row = rowbase + m;
    float ph = 0, px = 0;
#pragma unroll
    for (int j = 0; j < 8; ++j){ int c = lane + 32*j;
      hxv[ri][j] = hf[row*256 + c]; ph += hxv[ri][j]*hxv[ri][j];  // L2-hot re-touch
      float xv = xf[row*256 + c];  px += xv*xv; }
    hnv[ri] = sqrtf(wave_sum(ph));
    xnv[ri] = sqrtf(wave_sum(px));

    float yz[8], yr[8];
    // temp_z: W_hz,W_iz, bias[2], LN gamma/beta[0]
    gate_ln(&smem[1*4096 + m*256], &smem[3*4096 + m*256], bias+512, gam,     bet,     hnv[ri], xnv[ri], bz2, bzs, lane, yz);
    // temp_r: W_hr,W_ir, bias[0], LN gamma/beta[1]
    gate_ln(&smem[0*4096 + m*256], &smem[2*4096 + m*256], bias,     gam+256, bet+256, hnv[ri], xnv[ri], br2, brs, lane, yr);

    float wv[8]; float pw = 0;
#pragma unroll
    for (int j = 0; j < 8; ++j){
      zv[ri][j] = sigmoidf(yz[j]);
      float rv = sigmoidf(yr[j]);
      wv[j] = rv * hxv[ri][j];                     // r_t * hx
      pw += wv[j]*wv[j];
    }
    float wn_r = sqrtf(wave_sum(pw));
    float wn = fmaxf(wn_r, 1e-7f);
    float hn_c = fmaxf(hnv[ri], 1e-7f);
    float crh = (wn_r <= 1e-7f) ? 0.f : tanhf(wn/hn_c * artanh_pos(hn_c)) / wn;
#pragma unroll
    for (int j = 0; j < 8; ++j) rhv[ri][j] = crh * wv[j];
    rhn_[ri] = crh * wn_r;                         // ||rh_t||
  }
  __syncthreads();                                 // all gate-region reads done

  // re-stage rh_t as bf16 A-matrix in region 0 (8448 B < 16 KB region)
  unsigned short* sRh = (unsigned short*)smem;
#pragma unroll
  for (int ri = 0; ri < 2; ++ri)
#pragma unroll
    for (int j = 0; j < 8; ++j)
      sRh[(wave*2 + ri)*APITCH + lane + 32*j] = f2bf(rhv[ri][j]);
  __syncthreads();

  // ---- Phase B: mh_h = rh @ W_hh_^T ----
  v8f acc2[2] = {};
  const unsigned short* rowAr = sRh + lr*APITCH;
  for (int kb = 0; kb < 256; kb += 32){
    AB Ar;
    const int ka = kb + half*16;
    Ar.u[0] = *(const v4u*)(rowAr + ka); Ar.u[1] = *(const v4u*)(rowAr + ka + 8);
    const int kB = kb + half*16;
#pragma unroll
    for (int t = 0; t < 2; ++t){
      const int n = wave*32 + t*16 + lr;
      AB Bh; const unsigned short* p;
      p = whh + (size_t)(256 + n)*256 + kB; Bh.u[0]=*(const v4u*)p; Bh.u[1]=*(const v4u*)(p+8);
      acc2[t] = WMMA_BF16(Ar, Bh, acc2[t]);
    }
  }
  // spill mh_h -> region 1, mx_h -> region 2 (disjoint from live rh staging in region 0)
#pragma unroll
  for (int t = 0; t < 2; ++t)
#pragma unroll
    for (int v = 0; v < 8; ++v){
      smem[1*4096 + (half*8 + v)*256 + wave*32 + t*16 + lr] = acc2[t][v];
      smem[2*4096 + (half*8 + v)*256 + wave*32 + t*16 + lr] = accH[t][v];
    }
  __syncthreads();

  // ---- Epilogue B: h_tilde, final mobius combine ----
  float bh2, bhs;
  { float p2=0, ps=0;
#pragma unroll
    for (int j = 0; j < 8; ++j){ int c = lane + 32*j;
      float b = bias[256 + c]; p2 += b*b; ps += b; }
    bh2 = wave_sum(p2); bhs = wave_sum(ps);
  }

  for (int ri = 0; ri < 2; ++ri){
    const int m = wave*2 + ri;
    const size_t row = rowbase + m;

    float y[8];
    // h_tilde pre-activation: W_hh_,W_ih_, bias[1], LN gamma/beta[2]
    gate_ln(&smem[1*4096 + m*256], &smem[2*4096 + m*256], bias+256, gam+512, bet+512,
            rhn_[ri], xnv[ri], bh2, bhs, lane, y);

    float tv[8]; float pt = 0;
#pragma unroll
    for (int j = 0; j < 8; ++j){ tv[j] = tanhf(y[j]); pt += tv[j]*tv[j]; }
    float tn_r = sqrtf(wave_sum(pt));
    float tn = fmaxf(tn_r, 1e-7f);
    float che = tanhf(tn)/tn;                      // expmap0 scale
    float htn = tanhf(tn);                         // ||h_tilde||

    float w1[8], w2[8]; float p1=0, p2=0, p12=0;
#pragma unroll
    for (int j = 0; j < 8; ++j){
      float z = zv[ri][j];
      w1[j] = (1.f - z)*hxv[ri][j];                // (1-z)*hx
      w2[j] = z*che*tv[j];                         // z*h_tilde
      p1 += w1[j]*w1[j]; p2 += w2[j]*w2[j]; p12 += w1[j]*w2[j];
    }
    float s1 = wave_sum(p1), s2 = wave_sum(p2), s12 = wave_sum(p12);
    float hn_c = fmaxf(hnv[ri], 1e-7f);
    float w1n_r = sqrtf(s1), w2n_r = sqrtf(s2);
    float w1n = fmaxf(w1n_r, 1e-7f), w2n = fmaxf(w2n_r, 1e-7f);
    float c1 = (w1n_r <= 1e-7f) ? 0.f : tanhf(w1n/hn_c * artanh_pos(hn_c)) / w1n;
    float htn_c = fmaxf(htn, 1e-7f);
    float c2 = (w2n_r <= 1e-7f) ? 0.f : tanhf(w2n/htn_c * artanh_pos(htn_c)) / w2n;
    // final mobius_add(temp1, temp2)
    float a2 = c1*c1*s1, b2 = c2*c2*s2, ab = c1*c2*s12;
    float den = fmaxf(1.f + 2.f*ab + a2*b2, 1e-15f);
    float u = (1.f + 2.f*ab + b2)/den, v = (1.f - a2)/den;
#pragma unroll
    for (int j = 0; j < 8; ++j){ int c = lane + 32*j;
      out[row*256 + c] = u*c1*w1[j] + v*c2*w2[j];
    }
  }
}

extern "C" void kernel_launch(void* const* d_in, const int* in_sizes, int n_in,
                              void* d_out, int out_size, void* d_ws, size_t ws_size,
                              hipStream_t stream)
{
  const float* xf   = (const float*)d_in[0];
  const float* hf   = (const float*)d_in[1];
  const float* wih  = (const float*)d_in[2];
  const float* whh  = (const float*)d_in[3];
  const float* bias = (const float*)d_in[4];
  const float* gam  = (const float*)d_in[5];
  const float* bet  = (const float*)d_in[6];
  float* out = (float*)d_out;
  (void)n_in; (void)out_size; (void)ws_size;

  const size_t B = (size_t)in_sizes[0] / 256;

  // workspace: only bf16 weight copies (768 KB, L2-resident across all blocks)
  unsigned short* wihb = (unsigned short*)d_ws;
  unsigned short* whhb = wihb + 768*256;

  k_cvt<<<64, 256, 0, stream>>>(wih, wihb, 768L*256);
  k_cvt<<<64, 256, 0, stream>>>(whh, whhb, 768L*256);

  dim3 grid((unsigned)(B/16)), blk(256);
  k_mobius_gru<<<grid, blk, 0, stream>>>(wihb, whhb, xf, hf, bias, gam, bet, out);
}
